// ScaledDotProductAttention_25280177504393
// MI455X (gfx1250) — matrix operand order repaired and checked by reading, not compiled, not run on hardware
//
#include <hip/hip_runtime.h>

// Flash-attention forward, causal, B=4 H=16 S=2048 D=64, fp32 in/out.
// f16 WMMA (v_wmma_f32_16x16x32_f16), fp32 accum, online softmax (exp2-based),
// 64-key tiles, register double-buffered K/V staging, transposed V in LDS so
// all fragments load as aligned 16B LDS vectors. Row sums of P are computed on
// the XDL pipe as P x ones (2 extra WMMAs) with the running-length update
// folded into the WMMA C operand. Exponentials use the raw hardware v_exp_f32
// (flush-to-zero semantics are exactly right for decayed softmax terms).

typedef __attribute__((ext_vector_type(16))) _Float16 v16h;
typedef __attribute__((ext_vector_type(8)))  _Float16 v8h;
typedef __attribute__((ext_vector_type(8)))  float    v8f;

#define SLEN   2048
#define DH     64
#define QT     16            // q rows per wave
#define NWAVE  8
#define QTILE  128           // q rows per workgroup
#define KVT    64            // keys per iteration
#define KSTR   72            // halfs per row of Kl   (144B, 16B aligned)
#define VSTR   72            // halfs per row of Vt   (144B, 16B aligned)
#define PSTR   72            // halfs per row of P    (144B, 16B aligned)

#define WMMA_F16(A, B, C) \
    __builtin_amdgcn_wmma_f32_16x16x32_f16(false, (A), false, (B), (short)0, (C), false, false)

#if __has_builtin(__builtin_amdgcn_exp2f)
#define FAST_EXP2(x) __builtin_amdgcn_exp2f(x)   // bare v_exp_f32 (TRANS pipe)
#else
#define FAST_EXP2(x) __builtin_exp2f(x)
#endif

#if __has_builtin(__builtin_amdgcn_rcpf)
#define FAST_RCP(x) __builtin_amdgcn_rcpf(x)     // bare v_rcp_f32, ~1 ulp
#else
#define FAST_RCP(x) (1.0f / (x))
#endif

static __device__ __forceinline__ v16h ld_frag16(const _Float16* p) {
    // 16 consecutive halfs, 16B-aligned: two v8h loads
    v8h lo = *(const v8h*)p;
    v8h hi = *(const v8h*)(p + 8);
    return __builtin_shufflevector(lo, hi, 0,1,2,3,4,5,6,7,8,9,10,11,12,13,14,15);
}

static __device__ __forceinline__ v16h ld_afrag(const _Float16* p) {
    // A-layout: i<8 -> p[0..7], i>=8 -> p[16..23]
    v8h lo = *(const v8h*)p;
    v8h hi = *(const v8h*)(p + 16);
    return __builtin_shufflevector(lo, hi, 0,1,2,3,4,5,6,7,8,9,10,11,12,13,14,15);
}

__global__ __launch_bounds__(256)
void fa_fwd_kernel(const float* __restrict__ Q, const float* __restrict__ K,
                   const float* __restrict__ V, const int* __restrict__ maskp,
                   float* __restrict__ O)
{
    __shared__ _Float16 Kl[KVT * KSTR];        // [key][d]
    __shared__ _Float16 Vt[DH * VSTR];         // [d][key]  (transposed)
    __shared__ _Float16 Pl[NWAVE * QT * PSTR]; // per-wave P tiles [row][key]

    const int tid  = threadIdx.x;
    const int lane = tid & 31;
    const int wave = tid >> 5;
    const int nn   = lane & 15;   // A row / B,C,D column within 16
    const int kh   = lane >> 4;   // lane half

    const int qblk = blockIdx.x;
    const int bh   = blockIdx.y;
    const size_t base = (size_t)bh * (SLEN * DH);
    const int qbase   = qblk * QTILE + wave * QT;
    const int to_mask = maskp[0];

    // ---- Q -> two 16x32 f16 A-fragments; fold 1/sqrt(64)*log2(e) ----
    v16h aq0, aq1;
    {
        const float sc = 0.125f * 1.44269504088896340736f;
        const float* qrow = Q + base + (size_t)(qbase + nn) * DH;
        #pragma unroll
        for (int i = 0; i < 16; ++i) {
            const int k = i + kh * 16;
            aq0[i] = (_Float16)(sc * qrow[k]);
            aq1[i] = (_Float16)(sc * qrow[32 + k]);
        }
    }

    // constant ones B-matrix for row-sum WMMAs
    v16h ones;
    #pragma unroll
    for (int i = 0; i < 16; ++i) ones[i] = (_Float16)1.0f;

    v8f acc0 = {}, acc1 = {}, acc2 = {}, acc3 = {};
    v8f rowl = {};
    float rowm[8];
    #pragma unroll
    for (int r = 0; r < 8; ++r) rowm[r] = -__builtin_inff();

    const int qlast  = qbase + QT - 1;
    const int ntiles = to_mask ? (qblk + 1) * (QTILE / KVT) : (SLEN / KVT);

    // staging mapping: thread -> (key row r0, 16-col segment c0)
    const int r0 = tid & 63;
    const int c0 = (tid >> 6) << 4;

    // ---- prefetch tile 0 into registers ----
    float4 rk[4], rv[4];
    {
        const float* kp = K + base + (size_t)r0 * DH + c0;
        const float* vp = V + base + (size_t)r0 * DH + c0;
        #pragma unroll
        for (int j = 0; j < 4; ++j) {
            rk[j] = ((const float4*)kp)[j];
            rv[j] = ((const float4*)vp)[j];
        }
    }

    _Float16* pw = &Pl[wave * QT * PSTR];

    for (int t = 0; t < ntiles; ++t) {
        const int kv0 = t * KVT;

        __syncthreads();   // previous tile's readers done
        {
            // K: row-major f16, two aligned v8h stores
            v8h h0, h1;
            h0[0]=(_Float16)rk[0].x; h0[1]=(_Float16)rk[0].y; h0[2]=(_Float16)rk[0].z; h0[3]=(_Float16)rk[0].w;
            h0[4]=(_Float16)rk[1].x; h0[5]=(_Float16)rk[1].y; h0[6]=(_Float16)rk[1].z; h0[7]=(_Float16)rk[1].w;
            h1[0]=(_Float16)rk[2].x; h1[1]=(_Float16)rk[2].y; h1[2]=(_Float16)rk[2].z; h1[3]=(_Float16)rk[2].w;
            h1[4]=(_Float16)rk[3].x; h1[5]=(_Float16)rk[3].y; h1[6]=(_Float16)rk[3].z; h1[7]=(_Float16)rk[3].w;
            *(v8h*)&Kl[r0 * KSTR + c0]     = h0;
            *(v8h*)&Kl[r0 * KSTR + c0 + 8] = h1;
            // V: transpose into Vt[d][key] (scatter b16 stores, paid once/tile)
            #pragma unroll
            for (int j = 0; j < 4; ++j) {
                Vt[(c0 + j*4 + 0) * VSTR + r0] = (_Float16)rv[j].x;
                Vt[(c0 + j*4 + 1) * VSTR + r0] = (_Float16)rv[j].y;
                Vt[(c0 + j*4 + 2) * VSTR + r0] = (_Float16)rv[j].z;
                Vt[(c0 + j*4 + 3) * VSTR + r0] = (_Float16)rv[j].w;
            }
        }
        __syncthreads();

        // ---- issue next tile's global loads (latency hidden behind WMMAs) ----
        if (t + 1 < ntiles) {   // workgroup-uniform
            const float* kp = K + base + (size_t)(kv0 + KVT + r0) * DH + c0;
            const float* vp = V + base + (size_t)(kv0 + KVT + r0) * DH + c0;
            #pragma unroll
            for (int j = 0; j < 4; ++j) {
                rk[j] = ((const float4*)kp)[j];
                rv[j] = ((const float4*)vp)[j];
            }
        }

        if (!to_mask || kv0 <= qlast) {   // wave-uniform -> EXEC all-1s for WMMA
            // ---- scores: 4 chunks of 16 keys, D=64 via two WMMAs each ----
            v8f s[4];
            #pragma unroll
            for (int c = 0; c < 4; ++c) {
                const _Float16* kr = &Kl[(c * 16 + nn) * KSTR + kh * 16];
                v8f z = {};
                z = WMMA_F16(aq0, ld_frag16(kr), z);
                z = WMMA_F16(aq1, ld_frag16(kr + 32), z);
                s[c] = z;
            }

            // ---- causal mask: only the diagonal tile needs it ----
            if (to_mask && (kv0 + KVT - 1 > qbase)) {
                const int q0 = qbase + kh * 8;
                #pragma unroll
                for (int c = 0; c < 4; ++c) {
                    const int kg = kv0 + c * 16 + nn;
                    #pragma unroll
                    for (int r = 0; r < 8; ++r)
                        if (kg > q0 + r) s[c][r] = -__builtin_inff();
                }
            }

            // ---- row max (reduce across 16-lane half: rows stay in-half) ----
            float tm[8];
            #pragma unroll
            for (int r = 0; r < 8; ++r)
                tm[r] = fmaxf(fmaxf(s[0][r], s[1][r]), fmaxf(s[2][r], s[3][r]));
            #pragma unroll
            for (int off = 8; off; off >>= 1) {
                #pragma unroll
                for (int r = 0; r < 8; ++r)
                    tm[r] = fmaxf(tm[r], __shfl_xor(tm[r], off, 32));
            }

            // ---- online softmax (base-2): rescale, exponentiate, write P ----
            float scf[8];
            #pragma unroll
            for (int r = 0; r < 8; ++r) {
                const float mn = fmaxf(rowm[r], tm[r]);
                scf[r] = FAST_EXP2(rowm[r] - mn);   // 0 on first live tile
                rowm[r] = mn;
                const float p0 = FAST_EXP2(s[0][r] - mn);
                const float p1 = FAST_EXP2(s[1][r] - mn);
                const float p2 = FAST_EXP2(s[2][r] - mn);
                const float p3 = FAST_EXP2(s[3][r] - mn);
                _Float16* pr = pw + (r + 8 * kh) * PSTR + nn;
                pr[0]  = (_Float16)p0;
                pr[16] = (_Float16)p1;
                pr[32] = (_Float16)p2;
                pr[48] = (_Float16)p3;
                acc0[r] *= scf[r]; acc1[r] *= scf[r];
                acc2[r] *= scf[r]; acc3[r] *= scf[r];
            }

            // ---- P as two 16x32 A-fragments (LDS does C->A transpose) ----
            const v16h ap0 = ld_frag16(pw + nn * PSTR + kh * 16);
            const v16h ap1 = ld_frag16(pw + nn * PSTR + 32 + kh * 16);

            // ---- running length on the XDL pipe: rowl = rowl*scf + P.1 ----
            {
                v8f lfrag;
                #pragma unroll
                for (int r = 0; r < 8; ++r) lfrag[r] = rowl[r] * scf[r];
                lfrag = WMMA_F16(ap0, ones, lfrag);
                lfrag = WMMA_F16(ap1, ones, lfrag);
                rowl = lfrag;   // every lane holds its rows' sums (all cols equal)
            }

            // ---- O += P . V : 4 d-chunks x 2 key-halves ----
            {
                const _Float16* v0 = &Vt[nn * VSTR + kh * 16];           // dc=0
                const _Float16* v1 = &Vt[(16 + nn) * VSTR + kh * 16];    // dc=1
                const _Float16* v2 = &Vt[(32 + nn) * VSTR + kh * 16];    // dc=2
                const _Float16* v3 = &Vt[(48 + nn) * VSTR + kh * 16];    // dc=3
                acc0 = WMMA_F16(ap0, ld_frag16(v0), acc0);
                acc0 = WMMA_F16(ap1, ld_frag16(v0 + 32), acc0);
                acc1 = WMMA_F16(ap0, ld_frag16(v1), acc1);
                acc1 = WMMA_F16(ap1, ld_frag16(v1 + 32), acc1);
                acc2 = WMMA_F16(ap0, ld_frag16(v2), acc2);
                acc2 = WMMA_F16(ap1, ld_frag16(v2 + 32), acc2);
                acc3 = WMMA_F16(ap0, ld_frag16(v3), acc3);
                acc3 = WMMA_F16(ap1, ld_frag16(v3 + 32), acc3);
            }
        }
    }

    // ---- normalize and store O (fp32) ----
    #pragma unroll
    for (int r = 0; r < 8; ++r) {
        const float inv = FAST_RCP(rowl[r]);
        const size_t o = base + (size_t)(qbase + r + 8 * kh) * DH;
        O[o + nn]      = acc0[r] * inv;
        O[o + 16 + nn] = acc1[r] * inv;
        O[o + 32 + nn] = acc2[r] * inv;
        O[o + 48 + nn] = acc3[r] * inv;
    }
}

extern "C" void kernel_launch(void* const* d_in, const int* in_sizes, int n_in,
                              void* d_out, int out_size, void* d_ws, size_t ws_size,
                              hipStream_t stream) {
    const float* Q    = (const float*)d_in[0];
    const float* K    = (const float*)d_in[1];
    const float* V    = (const float*)d_in[2];
    const int*   mask = (const int*)d_in[3];
    float*       O    = (float*)d_out;

    dim3 grid(SLEN / QTILE, 4 * 16);   // 16 q-blocks x 64 (batch*head)
    fa_fwd_kernel<<<grid, 256, 0, stream>>>(Q, K, V, mask, O);
}
